// Qwen3Attention_38955353374863
// MI455X (gfx1250) — matrix operand order repaired and checked by reading, not compiled, not run on hardware
//
#include <hip/hip_runtime.h>

// ---------------------------------------------------------------------------
// Qwen3 attention block for MI455X (gfx1250, wave32, WMMA bf16 16x16x32).
//   S=2048, HID=4096, H=32, KVH=8, D=128.  All GEMMs via v_wmma_f32_16x16x32_bf16.
//   Attention K/V tiles staged with GLOBAL_LOAD_ASYNC_TO_LDS when available.
// ---------------------------------------------------------------------------

typedef __attribute__((ext_vector_type(16))) __bf16 v16bf;
typedef __attribute__((ext_vector_type(8)))  __bf16 v8bf;
typedef __attribute__((ext_vector_type(8)))  float  v8f;
typedef __attribute__((ext_vector_type(4)))  int    v4i;

typedef __attribute__((address_space(1))) v4i gas_v4i;   // global int4
typedef __attribute__((address_space(3))) v4i las_v4i;   // LDS int4

#define SEQ   2048
#define HID   4096
#define NH    32
#define NKVH  8
#define HD    128
#define QKVN  6144   // 4096 + 1024 + 1024

#if defined(__has_builtin)
#if __has_builtin(__builtin_amdgcn_global_load_async_to_lds_b128)
#define HAVE_ASYNC_LDS 1
#endif
#endif

__device__ inline v8f v8f_zero() {
    v8f z;
#pragma unroll
    for (int i = 0; i < 8; ++i) z[i] = 0.0f;
    return z;
}

__device__ inline v16bf bf_combine(v8bf lo, v8bf hi) {
    v16bf r;
#pragma unroll
    for (int i = 0; i < 8; ++i) { r[i] = lo[i]; r[i + 8] = hi[i]; }
    return r;
}

__device__ inline void wait_async_zero() {
#if defined(HAVE_ASYNC_LDS)
#if __has_builtin(__builtin_amdgcn_s_wait_asynccnt)
    __builtin_amdgcn_s_wait_asynccnt(0);
#else
    asm volatile("s_wait_asynccnt 0" ::: "memory");
#endif
#endif
}

// copy 16 bytes global -> LDS (async if supported, else through registers)
__device__ inline void copy16_g2l(const __bf16* __restrict__ src,
                                  __bf16* __restrict__ dst) {
#if defined(HAVE_ASYNC_LDS)
    __builtin_amdgcn_global_load_async_to_lds_b128(
        (gas_v4i*)(__bf16*)src, (las_v4i*)dst, 0, 0);
#else
    *(v8bf*)dst = *(const v8bf*)src;
#endif
}

// ---------------------------------------------------------------------------
// Kernel 1/4: tiled GEMM  C[M x N] (f32) = A[M x K] * B[K x N] (B is f32,
// converted to bf16 on the fly; A is f32 or bf16 per template).
// Block tile 128x128, K-step 32, 256 threads = 8 waves, each wave 32x64.
// ---------------------------------------------------------------------------
template <typename TA>
__global__ __launch_bounds__(256)
void gemm_bf16_kernel(const TA* __restrict__ A, int lda,
                      const float* __restrict__ B, int ldb,
                      float* __restrict__ C, int ldc, int K)
{
    constexpr int LDS_STRIDE = 40;                 // 32 + 8 pad (bf16 elems)
    __shared__ __bf16 ldsA[128 * LDS_STRIDE];      // 128 rows x 32 k
    __shared__ __bf16 ldsBt[128 * LDS_STRIDE];     // 128 n  x 32 k (transposed)

    const int tid   = threadIdx.x;
    const int lane  = tid & 31;
    const int wave  = tid >> 5;
    const int waveM = (wave & 3) * 32;
    const int waveN = (wave >> 2) * 64;
    const int tileM = blockIdx.y * 128;
    const int tileN = blockIdx.x * 128;

    const int m16  = lane & 15;
    const int kh   = lane >> 4;
    const int half = kh;

    v8f acc[2][4];
#pragma unroll
    for (int mi = 0; mi < 2; ++mi)
#pragma unroll
        for (int ni = 0; ni < 4; ++ni) acc[mi][ni] = v8f_zero();

    const int aRow = tid >> 1;             // 0..127
    const int aK0  = (tid & 1) * 16;       // 0 or 16
    const int bK   = tid >> 3;             // 0..31
    const int bN0  = (tid & 7) * 16;       // 0..112

    for (int kt = 0; kt < K; kt += 32) {
        __syncthreads();
        {   // stage A tile (convert to bf16)
            const TA* src = A + (size_t)(tileM + aRow) * lda + kt + aK0;
            __bf16* dst = &ldsA[aRow * LDS_STRIDE + aK0];
#pragma unroll
            for (int j = 0; j < 16; ++j) dst[j] = (__bf16)(float)src[j];
            if (kt + 32 < K) __builtin_prefetch(src + 32, 0, 1);
        }
        {   // stage B tile transposed: ldsBt[n][k]
            const float* src = B + (size_t)(kt + bK) * ldb + tileN + bN0;
#pragma unroll
            for (int j = 0; j < 16; ++j)
                ldsBt[(bN0 + j) * LDS_STRIDE + bK] = (__bf16)src[j];
            if (kt + 32 < K) __builtin_prefetch(src + (size_t)32 * ldb, 0, 1);
        }
        __syncthreads();

        // A fragments (16x32): lane m = m16; K = kh*8+i (i<8), 16+kh*8+i (i>=8)
        v16bf afrag[2];
#pragma unroll
        for (int mi = 0; mi < 2; ++mi) {
            const __bf16* p = &ldsA[(waveM + mi * 16 + m16) * LDS_STRIDE + kh * 8];
            afrag[mi] = bf_combine(*(const v8bf*)p, *(const v8bf*)(p + 16));
        }
#pragma unroll
        for (int ni = 0; ni < 4; ++ni) {
            // B fragment (32x16): lane n = m16; K = kh*8+i (i<8), 16+kh*8+i (i>=8)
            const __bf16* p = &ldsBt[(waveN + ni * 16 + m16) * LDS_STRIDE + kh * 8];
            v16bf bfrag = bf_combine(*(const v8bf*)p, *(const v8bf*)(p + 16));
#pragma unroll
            for (int mi = 0; mi < 2; ++mi)
                acc[mi][ni] = __builtin_amdgcn_wmma_f32_16x16x32_bf16(
                    false, afrag[mi], false, bfrag, (short)0, acc[mi][ni],
                    false, false);
        }
    }

    // C/D layout: row = r + 8*half, col = m16 (within each 16x16 frag)
#pragma unroll
    for (int mi = 0; mi < 2; ++mi)
#pragma unroll
        for (int ni = 0; ni < 4; ++ni)
#pragma unroll
            for (int r = 0; r < 8; ++r) {
                int m = tileM + waveM + mi * 16 + r + 8 * half;
                int n = tileN + waveN + ni * 16 + m16;
                C[(size_t)m * ldc + n] = acc[mi][ni][r];
            }
}

// ---------------------------------------------------------------------------
// Kernel 2/4: per-head RMSNorm + RoPE epilogue on the fused QKV output.
// One block (128 threads) per sequence position. Emits bf16:
//   Q[NH][S][D], K[NKVH][S][D], Vt[NKVH][D][S] (V transposed for attention).
// ---------------------------------------------------------------------------
__global__ __launch_bounds__(128)
void norm_rope_kernel(const float* __restrict__ qkv,
                      const float* __restrict__ cosb,
                      const float* __restrict__ sinb,
                      const float* __restrict__ qw,
                      const float* __restrict__ kw,
                      const float* __restrict__ epsp,
                      __bf16* __restrict__ Q,
                      __bf16* __restrict__ Kc,
                      __bf16* __restrict__ Vt)
{
    __shared__ float red[128];
    __shared__ float xs[128];
    const int s = blockIdx.x;
    const int d = threadIdx.x;
    const float eps = *epsp;
    const float cv = cosb[s * HD + d];
    const float sv = sinb[s * HD + d];
    const float* row = qkv + (size_t)s * QKVN;

#pragma unroll 1
    for (int h = 0; h < NH + NKVH; ++h) {
        const bool is_q = (h < NH);
        const int col = is_q ? (h * HD) : (HID + (h - NH) * HD);
        float x = row[col + d];
        red[d] = x * x;
        __syncthreads();
        for (int off = 64; off > 0; off >>= 1) {
            if (d < off) red[d] += red[d + off];
            __syncthreads();
        }
        float rinv = rsqrtf(red[0] * (1.0f / 128.0f) + eps);
        float xn = x * rinv * (is_q ? qw[d] : kw[d]);
        xs[d] = xn;
        __syncthreads();
        float rot = (d < 64) ? -xs[d + 64] : xs[d - 64];
        float y = xn * cv + rot * sv;
        if (is_q)
            Q[((size_t)h * SEQ + s) * HD + d] = (__bf16)y;
        else
            Kc[((size_t)(h - NH) * SEQ + s) * HD + d] = (__bf16)y;
        __syncthreads();
    }
    // V: no norm/rope, store transposed [kvh][d][s]
#pragma unroll 1
    for (int h = 0; h < NKVH; ++h) {
        float x = row[HID + NKVH * HD + h * HD + d];
        Vt[((size_t)h * HD + d) * SEQ + s] = (__bf16)x;
    }
}

// ---------------------------------------------------------------------------
// Kernel 3/4: flash-style attention (no mask, no scaling — matches reference).
// Grid: (S/128 query tiles, NH heads). 256 threads = 8 waves x 16 queries.
// Key tiles of 64, async LDS staging, online softmax, WMMA for QK^T and P*V.
// ---------------------------------------------------------------------------
__global__ __launch_bounds__(256)
void attn_kernel(const __bf16* __restrict__ Q,   // [NH][S][D]
                 const __bf16* __restrict__ Kc,  // [NKVH][S][D]
                 const __bf16* __restrict__ Vt,  // [NKVH][D][S]
                 __bf16* __restrict__ Out)       // [S][NH*D]
{
    constexpr int KT = 64;
    constexpr int LK = HD + 8;     // ldsK row stride (bf16)
    constexpr int LV = KT + 8;     // ldsVt row stride
    constexpr int LP = KT + 8;     // ldsP row stride
    __shared__ __bf16 ldsK[KT * LK];          // 64 keys x 128 d
    __shared__ __bf16 ldsVt[HD * LV];         // 128 d x 64 keys
    __shared__ __bf16 ldsP[8 * 16 * LP];      // per-wave 16 x 64 P tile

    const int head   = blockIdx.y;
    const int kvhead = head >> 2;             // NH / NKVH = 4
    const int q0     = blockIdx.x * 128;
    const int tid    = threadIdx.x;
    const int lane   = tid & 31;
    const int wave   = tid >> 5;
    const int wq0    = q0 + wave * 16;

    const int m16  = lane & 15;
    const int kh   = lane >> 4;
    const int half = kh;

    // Q fragments (held in registers for the whole K loop)
    v16bf qf[4];
    const __bf16* qbase = Q + ((size_t)head * SEQ + wq0 + m16) * HD;
#pragma unroll
    for (int f = 0; f < 4; ++f) {
        v8bf lo = *(const v8bf*)(qbase + f * 32 + kh * 8);
        v8bf hi = *(const v8bf*)(qbase + f * 32 + 16 + kh * 8);
        qf[f] = bf_combine(lo, hi);
    }

    v8f o[8];
#pragma unroll
    for (int f = 0; f < 8; ++f) o[f] = v8f_zero();
    float mrow[8], lrow[8];
#pragma unroll
    for (int r = 0; r < 8; ++r) { mrow[r] = -3.0e38f; lrow[r] = 0.0f; }

    const __bf16* kbase = Kc + (size_t)kvhead * SEQ * HD;
    const __bf16* vbase = Vt + (size_t)kvhead * HD * SEQ;

    for (int k0 = 0; k0 < SEQ; k0 += KT) {
        __syncthreads();
        {   // stage K tile: 64 keys x 128 d (pure bf16 copy -> async DMA)
            int r = tid >> 2, d0 = (tid & 3) * 32;
            const __bf16* src = kbase + (size_t)(k0 + r) * HD + d0;
            __bf16* dst = &ldsK[r * LK + d0];
#pragma unroll
            for (int j = 0; j < 32; j += 8) copy16_g2l(src + j, dst + j);
        }
        {   // stage Vt tile: 128 d x 64 keys
            int dd = tid >> 1, s0 = (tid & 1) * 32;
            const __bf16* src = vbase + (size_t)dd * SEQ + k0 + s0;
            __bf16* dst = &ldsVt[dd * LV + s0];
#pragma unroll
            for (int j = 0; j < 32; j += 8) copy16_g2l(src + j, dst + j);
        }
        wait_async_zero();
        __syncthreads();

        // ---- S = Q * K^T : 4 chunks of 16 keys, accumulate over d ----
        v8f sf[4];
#pragma unroll
        for (int n = 0; n < 4; ++n) {
            v8f accs = v8f_zero();
#pragma unroll
            for (int f = 0; f < 4; ++f) {
                const __bf16* p = &ldsK[(n * 16 + m16) * LK + f * 32 + kh * 8];
                v16bf bfrag = bf_combine(*(const v8bf*)p, *(const v8bf*)(p + 16));
                accs = __builtin_amdgcn_wmma_f32_16x16x32_bf16(
                    false, qf[f], false, bfrag, (short)0, accs, false, false);
            }
            sf[n] = accs;
        }

        // ---- online softmax over the 64 keys of this tile ----
        float nm[8], scale[8], rs[8];
#pragma unroll
        for (int r = 0; r < 8; ++r) {
            float v = mrow[r];
#pragma unroll
            for (int n = 0; n < 4; ++n) v = fmaxf(v, sf[n][r]);
#pragma unroll
            for (int off = 1; off < 16; off <<= 1)
                v = fmaxf(v, __shfl_xor(v, off, 32));
            nm[r] = v;
            scale[r] = __expf(mrow[r] - v);
            mrow[r] = v;
            rs[r] = 0.0f;
        }
#pragma unroll
        for (int n = 0; n < 4; ++n)
#pragma unroll
            for (int r = 0; r < 8; ++r) {
                float p = __expf(sf[n][r] - nm[r]);
                rs[r] += p;
                ldsP[(wave * 16 + r + 8 * half) * LP + n * 16 + m16] = (__bf16)p;
            }
#pragma unroll
        for (int r = 0; r < 8; ++r) {
            float t = rs[r];
#pragma unroll
            for (int off = 1; off < 16; off <<= 1)
                t += __shfl_xor(t, off, 32);
            lrow[r] = lrow[r] * scale[r] + t;
        }
#pragma unroll
        for (int f = 0; f < 8; ++f)
#pragma unroll
            for (int r = 0; r < 8; ++r) o[f][r] *= scale[r];

        // ---- O += P * V  (per-wave LDS round trip; same-wave LDS is in-order)
#pragma unroll
        for (int kc = 0; kc < 2; ++kc) {
            const __bf16* pp = &ldsP[(wave * 16 + m16) * LP + kc * 32 + kh * 8];
            v16bf pa = bf_combine(*(const v8bf*)pp, *(const v8bf*)(pp + 16));
#pragma unroll
            for (int f = 0; f < 8; ++f) {
                const __bf16* vp = &ldsVt[(f * 16 + m16) * LV + kc * 32 + kh * 8];
                v16bf bfrag = bf_combine(*(const v8bf*)vp, *(const v8bf*)(vp + 16));
                o[f] = __builtin_amdgcn_wmma_f32_16x16x32_bf16(
                    false, pa, false, bfrag, (short)0, o[f], false, false);
            }
        }
    }

    // normalize and store bf16 for the output projection
#pragma unroll
    for (int f = 0; f < 8; ++f)
#pragma unroll
        for (int r = 0; r < 8; ++r) {
            float val = o[f][r] / lrow[r];
            int row = wq0 + r + 8 * half;
            int col = head * HD + f * 16 + m16;
            Out[(size_t)row * (NH * HD) + col] = (__bf16)val;
        }
}

// ---------------------------------------------------------------------------
// Launch
// ---------------------------------------------------------------------------
extern "C" void kernel_launch(void* const* d_in, const int* in_sizes, int n_in,
                              void* d_out, int out_size, void* d_ws, size_t ws_size,
                              hipStream_t stream) {
    const float* hidden = (const float*)d_in[0];
    const float* cosb   = (const float*)d_in[1];
    const float* sinb   = (const float*)d_in[2];
    const float* wq     = (const float*)d_in[3];
    const float* wk     = (const float*)d_in[4];
    const float* wv     = (const float*)d_in[5];
    const float* wo     = (const float*)d_in[6];
    const float* qnw    = (const float*)d_in[7];
    const float* knw    = (const float*)d_in[8];
    const float* epsp   = (const float*)d_in[9];
    float* out          = (float*)d_out;

    char* ws = (char*)d_ws;
    size_t off = 0;
    float*  qkv  = (float*)(ws + off);  off += (size_t)SEQ * QKVN * 4;       // 48 MB
    __bf16* Qb   = (__bf16*)(ws + off); off += (size_t)NH * SEQ * HD * 2;    // 16 MB
    __bf16* Kb   = (__bf16*)(ws + off); off += (size_t)NKVH * SEQ * HD * 2;  //  4 MB
    __bf16* Vtb  = (__bf16*)(ws + off); off += (size_t)NKVH * HD * SEQ * 2;  //  4 MB
    __bf16* Attb = (__bf16*)(ws + off); off += (size_t)SEQ * NH * HD * 2;    // 16 MB

    dim3 blk(256);
    // QKV projections (fused into one [S x 6144] buffer, ldc = 6144)
    gemm_bf16_kernel<float><<<dim3(HID / 128, SEQ / 128), blk, 0, stream>>>(
        hidden, HID, wq, HID, qkv, QKVN, HID);
    gemm_bf16_kernel<float><<<dim3((NKVH * HD) / 128, SEQ / 128), blk, 0, stream>>>(
        hidden, HID, wk, NKVH * HD, qkv + HID, QKVN, HID);
    gemm_bf16_kernel<float><<<dim3((NKVH * HD) / 128, SEQ / 128), blk, 0, stream>>>(
        hidden, HID, wv, NKVH * HD, qkv + HID + NKVH * HD, QKVN, HID);

    // per-head RMSNorm + RoPE, bf16 repack
    norm_rope_kernel<<<dim3(SEQ), dim3(128), 0, stream>>>(
        qkv, cosb, sinb, qnw, knw, epsp, Qb, Kb, Vtb);

    // attention
    attn_kernel<<<dim3(SEQ / 128, NH), blk, 0, stream>>>(Qb, Kb, Vtb, Attb);

    // output projection -> fp32 result
    gemm_bf16_kernel<__bf16><<<dim3(HID / 128, SEQ / 128), blk, 0, stream>>>(
        Attb, NH * HD, wo, HID, out, HID, NH * HD);
}
